// AdaptivePropagation_15006615733587
// MI455X (gfx1250) — matrix operand order repaired and checked by reading, not compiled, not run on hardware
//
#include <hip/hip_runtime.h>

typedef __attribute__((ext_vector_type(16))) _Float16 v16h;
typedef __attribute__((ext_vector_type(8)))  _Float16 v8h;
typedef __attribute__((ext_vector_type(8)))  float    v8f;
typedef __attribute__((ext_vector_type(4)))  float    v4f;
typedef __attribute__((ext_vector_type(4)))  int      v4i;

#if defined(__AMDGCN__) && \
    __has_builtin(__builtin_amdgcn_global_load_async_to_lds_b128) && \
    __has_builtin(__builtin_amdgcn_s_wait_asynccnt)
#define USE_ASYNC_COPY 1
#else
#define USE_ASYNC_COPY 0
#endif

namespace {
constexpr int Bn  = 8;
constexpr int Kk  = 8;     // K offset pairs
constexpr int Hn  = 512;
constexpr int Wn  = 640;
constexpr int Cin = 32;
constexpr int C1  = 32;    // conv1 out channels
constexpr int C2  = 16;    // conv2 out channels (2K)
constexpr int TH  = 16;    // output tile height
constexpr int TW  = 32;    // output tile width
constexpr int FY  = TH + 4;  // 20 feature rows (halo 2)
constexpr int FX  = TW + 8;  // 40 feature cols (halo, x-quad aligned, origin x0-4)
constexpr int HY  = TH + 2;  // 18 hidden rows (halo 1)
constexpr int HX  = TW + 2;  // 34 hidden cols
constexpr int NT  = 256;     // 8 waves
}

__device__ __forceinline__ v16h cat8(v8h lo, v8h hi) {
  union { v16h v; v8h h[2]; } u;
  u.h[0] = lo; u.h[1] = hi;
  return u.v;
}

__global__ __launch_bounds__(NT)
void AdaptiveProp_fused(const float* __restrict__ depth,
                        const float* __restrict__ feat,
                        const float* __restrict__ w1,
                        const float* __restrict__ b1,
                        const float* __restrict__ w2,
                        const float* __restrict__ b2,
                        float* __restrict__ out)
{
  __shared__ alignas(32) _Float16 sF[FY][FX][Cin];   // 51200 B
  __shared__ alignas(32) _Float16 sH[HY][HX][C1];    // 39168 B
  __shared__ alignas(32) _Float16 sW1[9][C1][Cin];   // 18432 B
  __shared__ alignas(32) _Float16 sW2[9][C2][Cin];   //  9216 B
  __shared__ alignas(16) float    sD[TH * TW];       //  2048 B (depth tile)
  __shared__ float sB1[C1];
  __shared__ float sB2[C2];

  const int tid  = threadIdx.x;
  const int wave = tid >> 5;
  const int lane = tid & 31;       // wave32
  const int half = lane >> 4;
  const int lm   = lane & 15;

  const int x0 = blockIdx.x * TW;
  const int y0 = blockIdx.y * TH;
  const int b  = blockIdx.z;

  // ---- Stage 0a: kick off async depth-tile copy (overlaps the convs) --
#if USE_ASYNC_COPY
  if (tid < (TH * TW) / 4) {
    int yy = tid >> 3;             // 16 rows
    int xq = tid & 7;              // 8 quads of 4 floats per row
    const float* gp = depth + (b * Hn + (y0 + yy)) * Wn + x0 + 4 * xq;
    __builtin_amdgcn_global_load_async_to_lds_b128(
        (__attribute__((address_space(1))) v4i*)const_cast<float*>(gp),
        (__attribute__((address_space(3))) v4i*)&sD[tid * 4], 0, 0);
  }
#endif

  // ---- Stage 0b: cooperative loads into LDS ---------------------------
  // Feature halo tile: NCHW f32 -> [y][x][c] f16, zero padded.
  // x-origin = x0-4 so every float4 is 16B aligned and a quad is either
  // fully inside or fully outside the image in x (W and x0 are mult of 4).
  for (int i = tid; i < FY * (FX / 4) * Cin; i += NT) {
    int fxq = i % (FX / 4);
    int fy  = (i / (FX / 4)) % FY;
    int c   = i / ((FX / 4) * FY);
    int gy  = y0 - 2 + fy;
    int gx  = x0 - 4 + 4 * fxq;
    v4f q = {0.f, 0.f, 0.f, 0.f};
    if (gy >= 0 && gy < Hn && gx >= 0 && gx < Wn)
      q = *(const v4f*)(feat + ((b * Cin + c) * Hn + gy) * Wn + gx);
    _Float16* sp = &sF[fy][4 * fxq][c];
#pragma unroll
    for (int j = 0; j < 4; ++j) sp[j * Cin] = (_Float16)q[j];
  }
  // Weights (OIHW f32) -> [tap][n][cin] f16; float4 global reads.
  for (int i = tid; i < (9 * C1 * Cin) / 4; i += NT) {
    v4f q = *(const v4f*)(w1 + 4 * i);
#pragma unroll
    for (int j = 0; j < 4; ++j) {
      int e = 4 * i + j;
      int tap = e % 9, c = (e / 9) % Cin, n = e / (9 * Cin);
      sW1[tap][n][c] = (_Float16)q[j];
    }
  }
  for (int i = tid; i < (9 * C2 * Cin) / 4; i += NT) {
    v4f q = *(const v4f*)(w2 + 4 * i);
#pragma unroll
    for (int j = 0; j < 4; ++j) {
      int e = 4 * i + j;
      int tap = e % 9, c = (e / 9) % Cin, n = e / (9 * Cin);
      sW2[tap][n][c] = (_Float16)q[j];
    }
  }
  if (tid < C1) sB1[tid] = b1[tid];
  if (tid < C2) sB2[tid] = b2[tid];

#if !USE_ASYNC_COPY
  // Fallback: plain depth passthrough (channel 0).
  for (int i = tid; i < (TH * TW) / 4; i += NT) {
    int yy = i >> 3, xq = i & 7;
    *(v4f*)(out + ((b * (Kk + 1)) * Hn + (y0 + yy)) * Wn + x0 + 4 * xq) =
        *(const v4f*)(depth + (b * Hn + (y0 + yy)) * Wn + x0 + 4 * xq);
  }
#endif

  __syncthreads();

  // ---- Stage 1: conv1 (32->32, 3x3) + bias + ReLU -> sH ---------------
  // 54 jobs = 18 hidden rows x 3 M-tiles (x starts 0,16,18 cover [0,34)).
  const int xsArr[3] = {0, 16, 18};
  for (int j = wave; j < HY * 3; j += NT / 32) {
    int hy = j % HY;
    int xs = xsArr[j / HY];

    // Contiguous K layout: lanes 0-15 hold K {0..15},
    // lanes 16-31 hold K {16..31} -> one 32B run per lane.
    auto loadA = [&](int tap) -> v16h {
      int ky = tap / 3, kx = tap % 3;
      const _Float16* ap = &sF[hy + ky][xs + lm + kx + 2][0];
      return *(const v16h*)(ap + half * 16);
    };

    v8f acc0 = {};
    v8f acc1 = {};
    v16h aC = loadA(0);
#pragma unroll
    for (int tap = 0; tap < 9; ++tap) {
      v16h aN = (tap < 8) ? loadA(tap + 1) : aC;   // pipeline next A-frag
      v16h bB0 = *(const v16h*)&sW1[tap][lm][half * 16];
      v16h bB1 = *(const v16h*)&sW1[tap][16 + lm][half * 16];
      acc0 = __builtin_amdgcn_wmma_f32_16x16x32_f16(false, aC, false, bB0,
                                                    (short)0, acc0, false, false);
      acc1 = __builtin_amdgcn_wmma_f32_16x16x32_f16(false, aC, false, bB1,
                                                    (short)0, acc1, false, false);
      aC = aN;
    }
    int gy = y0 - 1 + hy;
    float bia0 = sB1[lm], bia1 = sB1[16 + lm];
#pragma unroll
    for (int r = 0; r < 8; ++r) {
      int hx = xs + 8 * half + r;          // M = 8*half + r
      int gx = x0 - 1 + hx;
      // zero the hidden halo outside the image (conv2 SAME padding)
      bool ok = (gy >= 0) & (gy < Hn) & (gx >= 0) & (gx < Wn);
      float v0 = ok ? fmaxf(acc0[r] + bia0, 0.f) : 0.f;
      float v1 = ok ? fmaxf(acc1[r] + bia1, 0.f) : 0.f;
      if (hx < HX) {
        sH[hy][hx][lm]      = (_Float16)v0;
        sH[hy][hx][16 + lm] = (_Float16)v1;
      }
    }
  }
  __syncthreads();

  // ---- Stage 2: conv2 (32->16, 3x3) -> offsets -> bilinear sample -----
  const float* dptr = depth + b * Hn * Wn;
  for (int j = wave; j < TH * 2; j += NT / 32) {
    int oy  = j >> 1;
    int xs2 = (j & 1) * 16;

    auto loadA = [&](int tap) -> v16h {
      int ky = tap / 3, kx = tap % 3;
      const _Float16* ap = &sH[oy + ky][xs2 + lm + kx][0];
      return *(const v16h*)(ap + half * 16);
    };

    v8f acc = {};
    v16h aC = loadA(0);
#pragma unroll
    for (int tap = 0; tap < 9; ++tap) {
      v16h aN = (tap < 8) ? loadA(tap + 1) : aC;
      v16h bB = *(const v16h*)&sW2[tap][lm][half * 16];
      acc = __builtin_amdgcn_wmma_f32_16x16x32_f16(false, aC, false, bB,
                                                   (short)0, acc, false, false);
      aC = aN;
    }
    float bia = sB2[lm];
    float mine[8], othr[8];
#pragma unroll
    for (int r = 0; r < 8; ++r) {
      mine[r] = acc[r] + bia;                 // my channel (N = lm)
      othr[r] = __shfl_xor(mine[r], 1, 32);   // partner channel (N ^ 1)
    }
    if ((lm & 1) == 0) {                      // even channel = x-offset lane
      int k  = lm >> 1;
      int gy = y0 + oy;
      int xb = x0 + xs2 + 8 * half;
      float s[8];
#pragma unroll
      for (int r = 0; r < 8; ++r) {
        float px = (float)(xb + r) + mine[r];
        float py = (float)gy + othr[r];
        px = fminf(fmaxf(px, 0.f), (float)(Wn - 1));
        py = fminf(fmaxf(py, 0.f), (float)(Hn - 1));
        float xf = floorf(px), yf = floorf(py);
        float wx = px - xf, wy = py - yf;
        int x0i = (int)xf, y0i = (int)yf;
        int x1i = min(x0i + 1, Wn - 1);
        int y1i = min(y0i + 1, Hn - 1);
        float v00 = dptr[y0i * Wn + x0i];
        float v01 = dptr[y0i * Wn + x1i];
        float v10 = dptr[y1i * Wn + x0i];
        float v11 = dptr[y1i * Wn + x1i];
        s[r] = (1.f - wy) * ((1.f - wx) * v00 + wx * v01) +
               wy * ((1.f - wx) * v10 + wx * v11);
      }
      float* op = out + ((b * (Kk + 1) + 1 + k) * Hn + gy) * Wn + xb;
      v4f s0 = {s[0], s[1], s[2], s[3]};
      v4f s1 = {s[4], s[5], s[6], s[7]};
      *(v4f*)op       = s0;
      *(v4f*)(op + 4) = s1;
    }
  }

#if USE_ASYNC_COPY
  // ---- Stage 3: drain async depth copy, write channel 0 ---------------
  __builtin_amdgcn_s_wait_asynccnt(0);
  if (tid < (TH * TW) / 4) {
    int yy = tid >> 3;
    int xq = tid & 7;
    *(v4f*)(out + ((b * (Kk + 1)) * Hn + (y0 + yy)) * Wn + x0 + 4 * xq) =
        *(const v4f*)&sD[tid * 4];
  }
#endif
}

extern "C" void kernel_launch(void* const* d_in, const int* in_sizes, int n_in,
                              void* d_out, int out_size, void* d_ws, size_t ws_size,
                              hipStream_t stream) {
  const float* depth = (const float*)d_in[0];
  const float* feat  = (const float*)d_in[1];
  const float* w1    = (const float*)d_in[2];
  const float* b1    = (const float*)d_in[3];
  const float* w2    = (const float*)d_in[4];
  const float* b2    = (const float*)d_in[5];
  float* out = (float*)d_out;
  (void)in_sizes; (void)n_in; (void)out_size; (void)d_ws; (void)ws_size;

  dim3 grid(Wn / TW, Hn / TH, Bn);   // 20 x 32 x 8 = 5120 workgroups
  AdaptiveProp_fused<<<grid, NT, 0, stream>>>(depth, feat, w1, b1, w2, b2, out);
}
